// GroupedQueryAttention_86191403696520
// MI455X (gfx1250) — compile-verified, matrix operand order checked by reading, not run on hardware
//
#include <hip/hip_runtime.h>
#include <hip/hip_bf16.h>

// ---------------------------------------------------------------------------
// GQA fused pipeline for gfx1250 (MI455X), bf16 WMMA + async LDS staging.
//   B=2, S=2048, HID=2048, H=32, D=64, KV=8, G=4
// ---------------------------------------------------------------------------

#define HID_  2048
#define H_    32
#define KV_   8
#define S_    2048
#define B_    2
#define D_    64
#define NTOK  (B_ * S_)        // 4096
#define KVDIM (KV_ * D_)       // 512

typedef __attribute__((ext_vector_type(16))) __bf16 v16bf;
typedef __attribute__((ext_vector_type(8)))  float  v8f;

union Frag {
    v16bf v;
    uint4 q[2];
    unsigned short s[16];
};

__device__ __forceinline__ unsigned short f2bf(float f) {
    unsigned int x = __float_as_uint(f);
    unsigned int r = x + 0x7FFFu + ((x >> 16) & 1u);   // round-nearest-even
    return (unsigned short)(r >> 16);
}

__device__ __forceinline__ v8f wmma_bf16(v16bf a, v16bf b, v8f c) {
    return __builtin_amdgcn_wmma_f32_16x16x32_bf16(
        /*neg_a=*/false, a, /*neg_b=*/false, b,
        /*c_mod=*/(short)0, c, /*reuse_a=*/false, /*reuse_b=*/false);
}

// CDNA5 async global -> LDS copy (no VGPR round-trip, tracked by ASYNCcnt).
// LDS dest address = low 32 bits of the flat shared pointer.
__device__ __forceinline__ void async_copy_b128(unsigned short* lds_dst,
                                                const unsigned short* gsrc) {
    unsigned ldsoff = (unsigned)(size_t)lds_dst;
    asm volatile("global_load_async_to_lds_b128 %0, %1, off"
                 :: "v"(ldsoff), "v"(gsrc)
                 : "memory");
}

__device__ __forceinline__ void wait_asynccnt0() {
    asm volatile("s_wait_asynccnt 0x0" ::: "memory");
}

// ---------------------------------------------------------------------------
// f32 -> bf16 conversion (vectorized x4)
// ---------------------------------------------------------------------------
__global__ void cvt_f32_bf16(const float* __restrict__ in,
                             unsigned short* __restrict__ out,
                             long long n4) {
    long long i = (long long)blockIdx.x * blockDim.x + threadIdx.x;
    long long stride = (long long)gridDim.x * blockDim.x;
    for (; i < n4; i += stride) {
        float4 f = ((const float4*)in)[i];
        uint2 pk;
        pk.x = (unsigned)f2bf(f.x) | ((unsigned)f2bf(f.y) << 16);
        pk.y = (unsigned)f2bf(f.z) | ((unsigned)f2bf(f.w) << 16);
        ((uint2*)out)[i] = pk;
    }
}

// ---------------------------------------------------------------------------
// Tiled bf16 GEMM:  out[m,n] = sum_k A[m,k] * W[n,k] + bias[n]
//   block tile 128x128, 8 waves (wave32), K-step 32.
//   Double-buffered LDS staged with global_load_async_to_lds_b128.
//   LDS pitch 40 elements (80B rows) -> conflict-free b128 ds loads.
// Store modes:
//   0: f32, plain [M,N]                 (final O projection -> d_out)
//   1: bf16 Q layout  [B][H][S][D]
//   2: bf16 K layout  [B][KV][S][D]
//   3: bf16 Vt layout [B][KV][D][S]     (V transposed for P@V B-fragments)
// ---------------------------------------------------------------------------
#define LDP 40

__global__ __launch_bounds__(256)
void gemm_bf16_wmma(const unsigned short* __restrict__ A,
                    const unsigned short* __restrict__ W,
                    const float* __restrict__ bias,
                    void* __restrict__ outp,
                    int M, int N, int K, int mode) {
    __shared__ unsigned short sA[2][128 * LDP];
    __shared__ unsigned short sB[2][128 * LDP];

    const int tid  = threadIdx.x;
    const int w    = tid >> 5;
    const int lane = tid & 31;
    const int lr   = lane & 15;
    const int kb   = (lane >> 4) << 3;   // {0,8}
    const int half = lane >> 4;
    const int row0 = blockIdx.y * 128;
    const int col0 = blockIdx.x * 128;
    const int wm   = w >> 2;             // 0..1  (64 rows each)
    const int wn   = w & 3;              // 0..3  (32 cols each)

    // per-thread staging chunks: 512 x 16B per (A,B) tile pair, 256 threads
    const int r0 = tid >> 2,         c0 = (tid & 3) << 3;
    const int r1 = (256 + tid) >> 2, c1 = ((256 + tid) & 3) << 3;

    auto stage = [&](int buf, int k0) {
        async_copy_b128(&sA[buf][r0 * LDP + c0],
                        &A[(size_t)(row0 + r0) * K + k0 + c0]);
        async_copy_b128(&sA[buf][r1 * LDP + c1],
                        &A[(size_t)(row0 + r1) * K + k0 + c1]);
        async_copy_b128(&sB[buf][r0 * LDP + c0],
                        &W[(size_t)(col0 + r0) * K + k0 + c0]);
        async_copy_b128(&sB[buf][r1 * LDP + c1],
                        &W[(size_t)(col0 + r1) * K + k0 + c1]);
    };

    v8f acc[4][2];
#pragma unroll
    for (int i = 0; i < 4; i++)
#pragma unroll
        for (int j = 0; j < 2; j++) acc[i][j] = 0.0f;

    // prologue: stage first tile
    stage(0, 0);
    wait_asynccnt0();
    __syncthreads();

    int buf = 0;
    for (int k0 = 0; k0 < K; k0 += 32) {
        if (k0 + 32 < K) stage(buf ^ 1, k0 + 32);   // overlap next tile

        Frag bfr[2];
#pragma unroll
        for (int nt = 0; nt < 2; nt++) {
            int rb = (wn * 32 + nt * 16 + lr) * LDP;
            bfr[nt].q[0] = *(const uint4*)&sB[buf][rb + kb];
            bfr[nt].q[1] = *(const uint4*)&sB[buf][rb + kb + 16];
        }
#pragma unroll
        for (int mt = 0; mt < 4; mt++) {
            Frag af;
            int ra = (wm * 64 + mt * 16 + lr) * LDP;
            af.q[0] = *(const uint4*)&sA[buf][ra + kb];
            af.q[1] = *(const uint4*)&sA[buf][ra + kb + 16];
            acc[mt][0] = wmma_bf16(af.v, bfr[0].v, acc[mt][0]);
            acc[mt][1] = wmma_bf16(af.v, bfr[1].v, acc[mt][1]);
        }

        wait_asynccnt0();     // my async copies into buf^1 have landed
        __syncthreads();      // everyone done reading buf / writing buf^1
        buf ^= 1;
    }

    // store:  lane L, reg r, tile(mt,nt) -> m = r + 8*half, n = lr
#pragma unroll
    for (int mt = 0; mt < 4; mt++)
#pragma unroll
        for (int nt = 0; nt < 2; nt++) {
            int n = col0 + wn * 32 + nt * 16 + lr;
            float bn = bias[n];
#pragma unroll
            for (int r = 0; r < 8; r++) {
                int m = row0 + wm * 64 + mt * 16 + r + 8 * half;
                float val = acc[mt][nt][r] + bn;
                if (mode == 0) {
                    ((float*)outp)[(size_t)m * N + n] = val;
                } else {
                    unsigned short hv = f2bf(val);
                    int bi = m / S_, s = m - bi * S_;
                    int hd = n >> 6, d = n & 63;
                    size_t idx;
                    if (mode == 1)        // Q: [B][H][S][D]
                        idx = (((size_t)(bi * H_ + hd)) * S_ + s) * D_ + d;
                    else if (mode == 2)   // K: [B][KV][S][D]
                        idx = (((size_t)(bi * KV_ + hd)) * S_ + s) * D_ + d;
                    else                  // Vt: [B][KV][D][S]
                        idx = (((size_t)(bi * KV_ + hd)) * D_ + d) * S_ + s;
                    ((unsigned short*)outp)[idx] = hv;
                }
            }
        }
}

// ---------------------------------------------------------------------------
// Flash attention: one wave per 16 queries, 32 keys per step.
//   scores = Q K^T * 0.125 + mask ; online softmax ; O += P V
// grid: (S/64, B*H), block: 128 (4 independent waves)
// ---------------------------------------------------------------------------
__global__ __launch_bounds__(128)
void attn_flash_wmma(const unsigned short* __restrict__ Qb,
                     const unsigned short* __restrict__ Kb,
                     const unsigned short* __restrict__ Vt,
                     const float* __restrict__ mask,
                     unsigned short* __restrict__ ctx) {
    __shared__ unsigned short sP[4 * 16 * LDP];   // wave-private P tiles

    const int w    = threadIdx.x >> 5;
    const int lane = threadIdx.x & 31;
    const int lr   = lane & 15;
    const int kb   = (lane >> 4) << 3;
    const int half = lane >> 4;

    const int bh = blockIdx.y;        // b*H + h
    const int b  = bh >> 5;
    const int h  = bh & 31;
    const int kv = h >> 2;            // G = 4
    const int q0 = blockIdx.x * 64 + w * 16;

    const unsigned short* Qrow = Qb + ((size_t)bh * S_ + q0) * D_;
    const unsigned short* Krow = Kb + ((size_t)(b * KV_ + kv) * S_) * D_;
    const unsigned short* Vrow = Vt + ((size_t)(b * KV_ + kv) * D_) * S_;
    const float* mrow = mask + (size_t)b * S_;
    unsigned short* sPw = sP + w * 16 * LDP;

    // Q fragments (d 0..31 and 32..63), reused across all key blocks
    Frag qf0, qf1;
    {
        const unsigned short* p = Qrow + (size_t)lr * D_;
        qf0.q[0] = *(const uint4*)&p[kb];
        qf0.q[1] = *(const uint4*)&p[kb + 16];
        qf1.q[0] = *(const uint4*)&p[32 + kb];
        qf1.q[1] = *(const uint4*)&p[32 + kb + 16];
    }

    v8f o[4];
#pragma unroll
    for (int i = 0; i < 4; i++) o[i] = 0.0f;
    float rm[8], rl[8];
#pragma unroll
    for (int r = 0; r < 8; r++) { rm[r] = -1e30f; rl[r] = 0.0f; }

    const float scale = 0.125f;       // 1/sqrt(64)

    for (int k0 = 0; k0 < S_; k0 += 32) {
        // ---- scores: two 16x16 tiles over 32 keys --------------------------
        v8f sc[2];
#pragma unroll
        for (int t = 0; t < 2; t++) {
            Frag kf0, kf1;
            const unsigned short* kp = Krow + (size_t)(k0 + t * 16 + lr) * D_;
            kf0.q[0] = *(const uint4*)&kp[kb];
            kf0.q[1] = *(const uint4*)&kp[kb + 16];
            kf1.q[0] = *(const uint4*)&kp[32 + kb];
            kf1.q[1] = *(const uint4*)&kp[32 + kb + 16];
            v8f z = 0.0f;
            z = wmma_bf16(qf0.v, kf0.v, z);
            z = wmma_bf16(qf1.v, kf1.v, z);
            sc[t] = z;
        }
        float mk0 = mrow[k0 + lr];
        float mk1 = mrow[k0 + 16 + lr];

        // ---- online softmax (row m = r + 8*half, cols across 16-lane half) -
        float alpha[8];
#pragma unroll
        for (int r = 0; r < 8; r++) {
            float a0 = sc[0][r] * scale + mk0;
            float a1 = sc[1][r] * scale + mk1;
            float mx = fmaxf(a0, a1);
            mx = fmaxf(mx, __shfl_xor(mx, 1));
            mx = fmaxf(mx, __shfl_xor(mx, 2));
            mx = fmaxf(mx, __shfl_xor(mx, 4));
            mx = fmaxf(mx, __shfl_xor(mx, 8));
            float mnew = fmaxf(rm[r], mx);
            float al = __expf(rm[r] - mnew);
            float p0 = __expf(a0 - mnew);
            float p1 = __expf(a1 - mnew);
            float sum = p0 + p1;
            sum += __shfl_xor(sum, 1);
            sum += __shfl_xor(sum, 2);
            sum += __shfl_xor(sum, 4);
            sum += __shfl_xor(sum, 8);
            rl[r] = rl[r] * al + sum;
            rm[r] = mnew;
            alpha[r] = al;
            sc[0][r] = p0;
            sc[1][r] = p1;
        }
#pragma unroll
        for (int dt = 0; dt < 4; dt++)
#pragma unroll
            for (int r = 0; r < 8; r++) o[dt][r] *= alpha[r];

        // ---- relayout P (D-layout -> A-fragment layout) through LDS --------
#pragma unroll
        for (int r = 0; r < 8; r++) {
            sPw[(r + 8 * half) * LDP + lr]      = f2bf(sc[0][r]);
            sPw[(r + 8 * half) * LDP + 16 + lr] = f2bf(sc[1][r]);
        }
        __syncthreads();
        Frag pf;
        pf.q[0] = *(const uint4*)&sPw[lr * LDP + kb];
        pf.q[1] = *(const uint4*)&sPw[lr * LDP + kb + 16];
        __syncthreads();

        // ---- O += P @ V (V columns are Vt rows, contiguous) ----------------
#pragma unroll
        for (int dt = 0; dt < 4; dt++) {
            Frag vf;
            const unsigned short* vp = Vrow + (size_t)(dt * 16 + lr) * S_ + k0;
            vf.q[0] = *(const uint4*)&vp[kb];
            vf.q[1] = *(const uint4*)&vp[kb + 16];
            o[dt] = wmma_bf16(pf.v, vf.v, o[dt]);
        }
    }

    // ---- normalize and store ctx[b][s][h*64 + d] as bf16 -------------------
#pragma unroll
    for (int dt = 0; dt < 4; dt++)
#pragma unroll
        for (int r = 0; r < 8; r++) {
            int s = q0 + r + 8 * half;
            float val = o[dt][r] / rl[r];
            ctx[((size_t)b * S_ + s) * HID_ + h * D_ + dt * 16 + lr] = f2bf(val);
        }
}

// ---------------------------------------------------------------------------
// Host orchestration
// ---------------------------------------------------------------------------
extern "C" void kernel_launch(void* const* d_in, const int* in_sizes, int n_in,
                              void* d_out, int out_size, void* d_ws, size_t ws_size,
                              hipStream_t stream) {
    const float* x    = (const float*)d_in[0];
    const float* mask = (const float*)d_in[1];
    const float* Wq   = (const float*)d_in[2];
    const float* bq   = (const float*)d_in[3];
    const float* Wk   = (const float*)d_in[4];
    const float* bk   = (const float*)d_in[5];
    const float* Wv   = (const float*)d_in[6];
    const float* bv   = (const float*)d_in[7];
    const float* Wo   = (const float*)d_in[8];
    const float* bo   = (const float*)d_in[9];

    // workspace layout (bytes)
    char* ws = (char*)d_ws;
    size_t off = 0;
    unsigned short* xb  = (unsigned short*)(ws + off); off += (size_t)NTOK * HID_ * 2;   // 16 MB
    unsigned short* wqb = (unsigned short*)(ws + off); off += (size_t)HID_ * HID_ * 2;   //  8 MB
    unsigned short* wkb = (unsigned short*)(ws + off); off += (size_t)KVDIM * HID_ * 2;  //  2 MB
    unsigned short* wvb = (unsigned short*)(ws + off); off += (size_t)KVDIM * HID_ * 2;  //  2 MB
    unsigned short* wob = (unsigned short*)(ws + off); off += (size_t)HID_ * HID_ * 2;   //  8 MB
    unsigned short* qb  = (unsigned short*)(ws + off); off += (size_t)NTOK * HID_ * 2;   // 16 MB
    unsigned short* kbf = (unsigned short*)(ws + off); off += (size_t)NTOK * KVDIM * 2;  //  4 MB
    unsigned short* vtb = (unsigned short*)(ws + off); off += (size_t)NTOK * KVDIM * 2;  //  4 MB
    unsigned short* ctx = (unsigned short*)(ws + off); off += (size_t)NTOK * HID_ * 2;   // 16 MB

    // 1) convert activations and weights to bf16
    cvt_f32_bf16<<<2048, 256, 0, stream>>>(x,  xb,  (long long)NTOK * HID_ / 4);
    cvt_f32_bf16<<<2048, 256, 0, stream>>>(Wq, wqb, (long long)HID_ * HID_ / 4);
    cvt_f32_bf16<<<1024, 256, 0, stream>>>(Wk, wkb, (long long)KVDIM * HID_ / 4);
    cvt_f32_bf16<<<1024, 256, 0, stream>>>(Wv, wvb, (long long)KVDIM * HID_ / 4);
    cvt_f32_bf16<<<2048, 256, 0, stream>>>(Wo, wob, (long long)HID_ * HID_ / 4);

    // 2) projections (Q/K/V) with attention-friendly output layouts
    gemm_bf16_wmma<<<dim3(HID_ / 128, NTOK / 128), 256, 0, stream>>>(
        xb, wqb, bq, qb, NTOK, HID_, HID_, /*mode=*/1);
    gemm_bf16_wmma<<<dim3(KVDIM / 128, NTOK / 128), 256, 0, stream>>>(
        xb, wkb, bk, kbf, NTOK, KVDIM, HID_, /*mode=*/2);
    gemm_bf16_wmma<<<dim3(KVDIM / 128, NTOK / 128), 256, 0, stream>>>(
        xb, wvb, bv, vtb, NTOK, KVDIM, HID_, /*mode=*/3);

    // 3) fused flash attention
    attn_flash_wmma<<<dim3(S_ / 64, B_ * H_), 128, 0, stream>>>(
        qb, kbf, vtb, mask, ctx);

    // 4) output projection -> f32 d_out
    gemm_bf16_wmma<<<dim3(HID_ / 128, NTOK / 128), 256, 0, stream>>>(
        ctx, wob, bo, d_out, NTOK, HID_, HID_, /*mode=*/0);
}
